// body_han1_9964324126985
// MI455X (gfx1250) — matrix operand order repaired and checked by reading, not compiled, not run on hardware
//
#include <hip/hip_runtime.h>
#include <hip/hip_bf16.h>

typedef __attribute__((ext_vector_type(16))) __bf16        v16bf;
typedef __attribute__((ext_vector_type(8)))  float         v8f;
typedef __attribute__((ext_vector_type(8)))  unsigned short us8;

#define BB 4
#define CC 256
#define HH 128
#define WW 128

// ---------- helpers ----------

__device__ __forceinline__ unsigned short f2bf(float f) {
    unsigned int u = __float_as_uint(f);
    u += 0x7FFFu + ((u >> 16) & 1u);
    return (unsigned short)(u >> 16);
}

union frag_cvt {
    struct { us8 lo, hi; } h;
    v16bf v;
};

// A-operand fragment from a K-major (row-major [M][K]) LDS tile.
// Lane layout (ISA 7.12.2): m = lane%16; elements 0..7 = K kb..kb+7,
// elements 8..15 = K 16+kb..16+kb+7, kb = (lane<16)?0:8.
// Both runs are contiguous -> two ds_load_b128.
__device__ __forceinline__ v16bf load_frag_a_vec(const unsigned short* base, int row0,
                                                 int k0, int stride, int lane) {
    const int m  = lane & 15;
    const int kb = (lane < 16) ? 0 : 8;
    const unsigned short* p = base + (size_t)(row0 + m) * stride + k0 + kb;
    frag_cvt f;
    f.h.lo = *(const us8*)(p);
    f.h.hi = *(const us8*)(p + 16);
    return f.v;
}

// B-operand fragment from a TRANSPOSED (row-major [N][K]) LDS tile.
// Lane layout: n = lane%16; elements 0..7 = K kb..kb+7,
// elements 8..15 = K 16+kb..16+kb+7, kb = (lane<16)?0:8 -> two ds_load_b128.
__device__ __forceinline__ v16bf load_frag_b_vec(const unsigned short* baseT, int k0,
                                                 int col0, int strideT, int lane) {
    const int n  = lane & 15;
    const int kb = (lane < 16) ? 0 : 8;
    const unsigned short* p = baseT + (size_t)(col0 + n) * strideT + k0 + kb;
    frag_cvt f;
    f.h.lo = *(const us8*)(p);
    f.h.hi = *(const us8*)(p + 16);
    return f.v;
}

// ---------- kernel 1: pooled means + encoder linears ----------
// Emits enc PRE-TRANSPOSED as [b][d][r] (K-major for the GEMM1 B-operand).

__global__ void __launch_bounds__(256)
pool_enc_kernel(const float* __restrict__ ex,
                const float* __restrict__ W1, const float* __restrict__ b1,
                const float* __restrict__ W2, const float* __restrict__ b2,
                unsigned short* __restrict__ encHT,
                unsigned short* __restrict__ encWT) {
    __shared__ float pooled[CC];
    const int r    = blockIdx.x;   // h (path 0) or w (path 1)
    const int b    = blockIdx.y;
    const int path = blockIdx.z;
    const int tid  = threadIdx.x;

    float s = 0.0f;
    if (path == 0) {
        const float* p = ex + (((size_t)b * CC + tid) * HH + r) * WW;
        for (int w = 0; w < WW; ++w) s += p[w];
    } else {
        const float* p = ex + (((size_t)b * CC + tid) * HH) * WW + r;
        for (int h = 0; h < HH; ++h) s += p[(size_t)h * WW];
    }
    pooled[tid] = s * (1.0f / 128.0f);
    __syncthreads();

    const float* Wm = (path == 0) ? W1 : W2;
    const float* bv = (path == 0) ? b1 : b2;
    float acc = bv[tid];
    const float* wrow = Wm + (size_t)tid * CC;
    for (int c = 0; c < CC; ++c) acc += pooled[c] * wrow[c];

    unsigned short* enc = (path == 0) ? encHT : encWT;
    enc[((size_t)b * CC + tid) * 128 + r] = f2bf(acc);   // [b][d][r]
}

// ---------- kernel 2: fused batched attention ----------
// GEMM1 (regs) -> in-register softmax (shfl) -> P to LDS -> GEMM2.

__global__ void __launch_bounds__(256)
attn_wmma_kernel(const float* __restrict__ body,
                 const unsigned short* __restrict__ encHT,
                 const unsigned short* __restrict__ encWT,
                 float* __restrict__ augH,
                 float* __restrict__ augW) {
    extern __shared__ char smem[];
    unsigned short* Ash = (unsigned short*)smem;              // feat   [256 c][128 r]  (A for GEMM1)
    unsigned short* Ath = (unsigned short*)(smem + 65536);    // feat^T [128 r][256 c]  (B for GEMM2)
    unsigned short* Bsh = (unsigned short*)(smem + 131072);   // enc^T  [256 d][128 k]  (B for GEMM1)
    unsigned short* Psh = (unsigned short*)(smem + 196608);   // P      [128 row][256 d] (A for GEMM2)

    const int half = blockIdx.x;
    const int i    = blockIdx.y;
    const int path = blockIdx.z;
    const int tid  = threadIdx.x;
    const int lane = tid & 31;
    const int wv   = tid >> 5;

    const int b   = i >> 7;
    const int fix = i & 127;

    // feat tile: dual-layout store (Ash row-major, Ath transposed).
    for (int e = tid; e < CC * 128; e += 256) {
        const int c = e >> 7, r = e & 127;
        float v = (path == 0)
            ? body[(((size_t)b * CC + c) * HH + r) * WW + fix]
            : body[(((size_t)b * CC + c) * HH + fix) * WW + r];
        const unsigned short bf = f2bf(v);
        Ash[e] = bf;
        Ath[(size_t)r * CC + c] = bf;
    }
    // enc^T tile: straight vectorized copy (global is already [d][k]).
    {
        const uint4* src = (const uint4*)(((path == 0) ? encHT : encWT)
                                          + (size_t)(i & 3) * CC * 128);
        uint4* dst = (uint4*)Bsh;
        for (int t = tid; t < (CC * 128) / 8; t += 256) dst[t] = src[t];
    }
    __syncthreads();

    // GEMM1: S rows [half*128 + wv*16 .. +16) x all 256 cols, K = 128. acc in regs.
    const int rowbase = half * 128;
    const int row0    = rowbase + wv * 16;
    v8f acc[16];
    v8f z = {};
#pragma unroll
    for (int t = 0; t < 16; ++t) acc[t] = z;

#pragma unroll
    for (int kk = 0; kk < 4; ++kk) {
        v16bf a = load_frag_a_vec(Ash, row0, kk * 32, 128, lane);
#pragma unroll
        for (int ct = 0; ct < 16; ++ct) {
            v16bf bm = load_frag_b_vec(Bsh, kk * 32, ct * 16, 128, lane);
            acc[ct] = __builtin_amdgcn_wmma_f32_16x16x32_bf16(
                false, a, false, bm, (short)0, acc[ct], false, false);
        }
    }

    // In-register softmax. Row m = j + 8*(lane>>4) of this wave's 16-row band lives
    // entirely within one 16-lane half; columns are spread over lanes (n) x tiles (ct).
#pragma unroll
    for (int j = 0; j < 8; ++j) {
        float mx = acc[0][j];
#pragma unroll
        for (int ct = 1; ct < 16; ++ct) mx = fmaxf(mx, acc[ct][j]);
#pragma unroll
        for (int m = 1; m < 16; m <<= 1) mx = fmaxf(mx, __shfl_xor(mx, m, 32));
        float s = 0.0f;
#pragma unroll
        for (int ct = 0; ct < 16; ++ct) {
            const float e = __expf(acc[ct][j] - mx);
            acc[ct][j] = e;
            s += e;
        }
#pragma unroll
        for (int m = 1; m < 16; m <<= 1) s += __shfl_xor(s, m, 32);
        const float inv = 1.0f / s;
        const int lrow = wv * 16 + j + ((lane >> 4) << 3);
#pragma unroll
        for (int ct = 0; ct < 16; ++ct)
            Psh[(size_t)lrow * CC + ct * 16 + (lane & 15)] = f2bf(acc[ct][j] * inv);
    }
    __syncthreads();

    // GEMM2: Out(128x128) = P(128x256) @ feat(256x128), K = 256.
    float* aug = (path == 0) ? augH : augW;
    v8f oacc[8];
#pragma unroll
    for (int t = 0; t < 8; ++t) oacc[t] = z;

#pragma unroll
    for (int kk = 0; kk < 8; ++kk) {
        v16bf a = load_frag_a_vec(Psh, wv * 16, kk * 32, 256, lane);
#pragma unroll
        for (int ct = 0; ct < 8; ++ct) {
            v16bf bm = load_frag_b_vec(Ath, kk * 32, ct * 16, 256, lane);
            oacc[ct] = __builtin_amdgcn_wmma_f32_16x16x32_bf16(
                false, a, false, bm, (short)0, oacc[ct], false, false);
        }
    }
#pragma unroll
    for (int ct = 0; ct < 8; ++ct) {
        const int n = ct * 16 + (lane & 15);
#pragma unroll
        for (int j = 0; j < 8; ++j) {
            const int cg = rowbase + wv * 16 + j + ((lane >> 4) << 3);
            aug[((size_t)i * CC + cg) * 128 + n] = oacc[ct][j];
        }
    }
}

// ---------- kernel 3: out = gamma*(aug_h + aug_w) + body -> bf16 conv input ----------

__global__ void __launch_bounds__(256)
combine_kernel(const float* __restrict__ augH, const float* __restrict__ augW,
               const float* __restrict__ body, const float* __restrict__ gamma,
               unsigned short* __restrict__ Xbf) {
    const size_t idx = (size_t)blockIdx.x * 256 + threadIdx.x;
    const int w = (int)(idx & 127);
    const int h = (int)((idx >> 7) & 127);
    const int c = (int)((idx >> 14) & 255);
    const int b = (int)(idx >> 22);
    const float g  = gamma[0];
    const float ah = augH[(((size_t)b * 128 + w) * CC + c) * 128 + h];
    const float aw = augW[(((size_t)b * 128 + h) * CC + c) * 128 + w];
    Xbf[idx] = f2bf(g * (ah + aw) + body[idx]);
}

// ---------- kernel 4: conv weight f32 -> bf16 ----------

__global__ void convert_w_kernel(const float* __restrict__ w,
                                 unsigned short* __restrict__ o, int n) {
    const int i = blockIdx.x * 256 + threadIdx.x;
    if (i < n) o[i] = f2bf(w[i]);
}

// ---------- kernel 5: 3x3 conv as implicit GEMM + BN + ReLU ----------
// M=256, N=65536 (one (b,y) row per N-tile), K=2304.

__global__ void __launch_bounds__(256)
conv_wmma_kernel(const unsigned short* __restrict__ Xbf,
                 const unsigned short* __restrict__ Wbf,
                 const float* __restrict__ bn_w,
                 const float* __restrict__ bn_b,
                 float* __restrict__ out) {
    __shared__ unsigned short Wt[128 * 32];  // [128 M][32 K], K-major (A operand)
    __shared__ unsigned short Xt[128 * 32];  // [128 N][32 K], K-major (B operand, transposed)

    const int ntile = blockIdx.x;
    const int m0    = blockIdx.y * 128;
    const int b     = ntile >> 7;
    const int y     = ntile & 127;
    const int tid   = threadIdx.x;
    const int lane  = tid & 31;
    const int wv    = tid >> 5;

    v8f acc[8];
    v8f z = {};
#pragma unroll
    for (int t = 0; t < 8; ++t) acc[t] = z;

    for (int kt = 0; kt < 72; ++kt) {
        const int k0 = kt * 32;
        // weight tile: vectorized copy, K contiguous
        {
            uint4* dw = (uint4*)Wt;
            for (int t = tid; t < 512; t += 256) {
                const int e  = t * 8;
                const int mm = e >> 5, kk = e & 31;
                dw[t] = *(const uint4*)(Wbf + (size_t)(m0 + mm) * 2304 + k0 + kk);
            }
        }
        if (kt + 1 < 72)
            __builtin_prefetch(&Wbf[(size_t)(m0 + (tid >> 1)) * 2304 + k0 + 32], 0, 1);
        // im2col tile, stored transposed [x][k]
        for (int e = tid; e < 128 * 32; e += 256) {
            const int kk = e & 31, xx = e >> 5;
            const int k   = k0 + kk;
            const int cin = k / 9;
            const int rem = k - cin * 9;
            const int ky  = rem / 3;
            const int kx  = rem - ky * 3;
            const int yy  = y + ky - 1;
            const int xs  = xx + kx - 1;
            unsigned short v = 0;
            if (yy >= 0 && yy < HH && xs >= 0 && xs < WW)
                v = Xbf[(((size_t)b * CC + cin) * HH + yy) * WW + xs];
            Xt[(size_t)xx * 32 + kk] = v;
        }
        __syncthreads();
        v16bf a = load_frag_a_vec(Wt, wv * 16, 0, 32, lane);
#pragma unroll
        for (int ct = 0; ct < 8; ++ct) {
            v16bf bm = load_frag_b_vec(Xt, 0, ct * 16, 32, lane);
            acc[ct] = __builtin_amdgcn_wmma_f32_16x16x32_bf16(
                false, a, false, bm, (short)0, acc[ct], false, false);
        }
        __syncthreads();
    }

    const float rs = rsqrtf(1.0f + 1e-5f);
#pragma unroll
    for (int ct = 0; ct < 8; ++ct) {
        const int x = ct * 16 + (lane & 15);
#pragma unroll
        for (int j = 0; j < 8; ++j) {
            const int m = m0 + wv * 16 + j + ((lane >> 4) << 3);
            float v = acc[ct][j] * (bn_w[m] * rs) + bn_b[m];
            out[(((size_t)b * CC + m) * HH + y) * WW + x] = fmaxf(v, 0.0f);
        }
    }
}

// ---------- launch ----------

extern "C" void kernel_launch(void* const* d_in, const int* in_sizes, int n_in,
                              void* d_out, int out_size, void* d_ws, size_t ws_size,
                              hipStream_t stream) {
    (void)in_sizes; (void)n_in; (void)out_size; (void)ws_size;
    const float* body  = (const float*)d_in[0];
    const float* ex    = (const float*)d_in[1];
    const float* W1    = (const float*)d_in[2];
    const float* b1    = (const float*)d_in[3];
    const float* W2    = (const float*)d_in[4];
    const float* b2    = (const float*)d_in[5];
    const float* convw = (const float*)d_in[6];
    const float* bn_w  = (const float*)d_in[7];
    const float* bn_b  = (const float*)d_in[8];
    const float* gamma = (const float*)d_in[9];
    float* out = (float*)d_out;

    char* ws = (char*)d_ws;
    unsigned short* encHT = (unsigned short*)(ws);                   // 256 KB
    unsigned short* encWT = (unsigned short*)(ws + (size_t)262144);  // 256 KB
    float* augH = (float*)(ws + (size_t)524288);                     // 64 MB
    float* augW = (float*)(ws + (size_t)524288 + 67108864);          // 64 MB
    unsigned short* Xbf =
        (unsigned short*)(ws + (size_t)524288 + 2ull * 67108864);    // 32 MB
    unsigned short* Wbf =
        (unsigned short*)(ws + (size_t)524288 + 2ull * 67108864 + 33554432);

    hipFuncSetAttribute((const void*)attn_wmma_kernel,
                        hipFuncAttributeMaxDynamicSharedMemorySize, 262144);

    pool_enc_kernel<<<dim3(128, 4, 2), 256, 0, stream>>>(ex, W1, b1, W2, b2, encHT, encWT);
    convert_w_kernel<<<(589824 + 255) / 256, 256, 0, stream>>>(convw, Wbf, 589824);
    attn_wmma_kernel<<<dim3(2, 512, 2), 256, 262144, stream>>>(body, encHT, encWT, augH, augW);
    combine_kernel<<<65536, 256, 0, stream>>>(augH, augW, body, gamma, Xbf);
    conv_wmma_kernel<<<dim3(512, 2), 256, 0, stream>>>(Xbf, Wbf, bn_w, bn_b, out);
}
